// G2MoEGRINMoeLayer_54863912239458
// MI455X (gfx1250) — matrix operand order repaired and checked by reading, not compiled, not run on hardware
//
#include <hip/hip_runtime.h>
#include <cmath>

// Problem dimensions (fixed by the reference)
#define Bc 2
#define Sc 2048
#define Hc 1024
#define Ic 4096
#define Ec 8
#define Tc (Bc * Sc)          // 4096 tokens
#define MAX_TILES (Tc / 16)   // 256 16-token tiles max per expert

typedef __bf16 v16bf  __attribute__((ext_vector_type(16)));
typedef __bf16 bf16x8 __attribute__((ext_vector_type(8)));
typedef float  v8f    __attribute__((ext_vector_type(8)));

// LDS layout (bf16 elements), padded row strides to avoid bank conflicts
#define XS_STRIDE 1032                 // 1024 + 8
#define AS_STRIDE 4104                 // 4096 + 8
#define XS_BYTES  (16 * XS_STRIDE * 2) // 33,024 B
#define AS_BYTES  (16 * AS_STRIDE * 2) // 131,328 B
#define SMEM_BYTES (XS_BYTES + AS_BYTES + 64)

__device__ __forceinline__ v8f wmma_bf16(v16bf a, v16bf b, v8f c) {
    // D = A(16x32 bf16) x B(32x16 bf16) + C(16x16 f32)
    return __builtin_amdgcn_wmma_f32_16x16x32_bf16(
        /*neg_a=*/false, a, /*neg_b=*/false, b,
        /*c_mod=*/(short)0, c, /*reuse_a=*/false, /*reuse_b=*/false);
}

// Branch-free tanh-gelu: gelu(g) = g * sigmoid(1.595769*(g + 0.044715*g^3))
// sigmoid(y) = rcp(1 + exp2(-log2(e)*y)); lowers to v_exp_f32 + v_rcp_f32,
// no EXEC-mask branches (unlike libm tanhf).
__device__ __forceinline__ float gelu_tanh(float g) {
    float y  = 1.5957691f * (g + 0.044715f * g * g * g);
    float e  = __builtin_amdgcn_exp2f(-1.4426950f * y);
    return g * __builtin_amdgcn_rcpf(1.0f + e);
}

// Load 16 consecutive fp32 (64B, aligned) and convert to a bf16 B-fragment.
// In this code's K order (shared with A), lane (l) of a 32x16 B-frag holds K = (l<16 ? 0..15 : 16..31)
// for column N = l&15 — i.e. 16 contiguous K values per lane.
__device__ __forceinline__ v16bf load_b_frag(const float* __restrict__ p) {
    const float4* q = (const float4*)p;
    float4 f0 = q[0], f1 = q[1], f2 = q[2], f3 = q[3];
    v16bf r;
    r[0]  = (__bf16)f0.x; r[1]  = (__bf16)f0.y; r[2]  = (__bf16)f0.z; r[3]  = (__bf16)f0.w;
    r[4]  = (__bf16)f1.x; r[5]  = (__bf16)f1.y; r[6]  = (__bf16)f1.z; r[7]  = (__bf16)f1.w;
    r[8]  = (__bf16)f2.x; r[9]  = (__bf16)f2.y; r[10] = (__bf16)f2.z; r[11] = (__bf16)f2.w;
    r[12] = (__bf16)f3.x; r[13] = (__bf16)f3.y; r[14] = (__bf16)f3.z; r[15] = (__bf16)f3.w;
    return r;
}

// A-fragment (16x32 bf16) from LDS. Lane l holds row M=l&15;
// lanes 0-15 hold K = 0..15; lanes 16-31 hold K = 16..31 (contiguous, matching
// B). So: two aligned 16B LDS loads at base + half*16 and base + half*16 + 8.
__device__ __forceinline__ v16bf load_a_frag(const __bf16* rowp /* = row base + half*16 */,
                                             int k0) {
    bf16x8 lo = *(const bf16x8*)(rowp + k0);
    bf16x8 hi = *(const bf16x8*)(rowp + k0 + 8);
    return __builtin_shufflevector(lo, hi, 0, 1, 2, 3, 4, 5, 6, 7,
                                           8, 9, 10, 11, 12, 13, 14, 15);
}

// ---------------------------------------------------------------------------
// Kernel 1: router. One thread per token: logits (fp32, exact), top-1 argmax
// (strict > keeps first max, matching jnp.argmax), build per-expert token
// lists via atomics. List order is non-deterministic but output is a
// per-token scatter, so results are deterministic.
// ---------------------------------------------------------------------------
__global__ void router_kernel(const float* __restrict__ x,
                              const float* __restrict__ gw,
                              float* __restrict__ logits,
                              int* __restrict__ cnt,
                              int* __restrict__ perm) {
    int t = blockIdx.x * blockDim.x + threadIdx.x;
    if (t >= Tc) return;
    const float4* xr = (const float4*)(x + (size_t)t * Hc);
    float acc[Ec];
#pragma unroll
    for (int e = 0; e < Ec; ++e) acc[e] = 0.0f;
    for (int h4 = 0; h4 < Hc / 4; ++h4) {
        float4 xv = xr[h4];
#pragma unroll
        for (int e = 0; e < Ec; ++e) {
            float4 gv = ((const float4*)(gw + (size_t)e * Hc))[h4];
            acc[e] += xv.x * gv.x + xv.y * gv.y + xv.z * gv.z + xv.w * gv.w;
        }
    }
    float best = acc[0];
    int be = 0;
#pragma unroll
    for (int e = 1; e < Ec; ++e)
        if (acc[e] > best) { best = acc[e]; be = e; }
#pragma unroll
    for (int e = 0; e < Ec; ++e) logits[(size_t)t * Ec + e] = acc[e];
    int slot = atomicAdd(&cnt[be], 1);
    perm[be * Tc + slot] = t;
}

// ---------------------------------------------------------------------------
// Kernel 2: fused expert FFN for one 16-token tile of one expert.
// Stage A: act[16, I] = gelu_tanh(x_tile @ Wg^T) * (x_tile @ Wu^T)  (LDS, bf16)
// Stage B: out[16, H] = act @ Wd^T, scattered to token rows (fp32)
// ---------------------------------------------------------------------------
__global__ void moe_expert_kernel(const float* __restrict__ x,
                                  const float* __restrict__ w_gate,
                                  const float* __restrict__ w_up,
                                  const float* __restrict__ w_down,
                                  float* __restrict__ out,
                                  const int* __restrict__ cnt,
                                  const int* __restrict__ perm) {
    extern __shared__ char smem[];
    __bf16* xs   = (__bf16*)smem;                       // [16][XS_STRIDE]
    __bf16* acts = (__bf16*)(smem + XS_BYTES);          // [16][AS_STRIDE]
    int*    toks = (int*)(smem + XS_BYTES + AS_BYTES);  // [16]

    const int e    = blockIdx.x >> 8;    // expert
    const int tile = blockIdx.x & 255;   // tile within expert
    const int nTok = cnt[e];
    if (tile * 16 >= nTok) return;       // block-uniform early exit
    int rem = nTok - tile * 16;
    if (rem > 16) rem = 16;

    // ---- load + convert the gathered 16xH x-tile into LDS (b128 stores) ----
    {
        const int row = threadIdx.x >> 4;          // 0..15
        const int c0  = (threadIdx.x & 15) * 64;   // 64 floats per thread
        int tok = (row < rem) ? perm[e * Tc + tile * 16 + row] : -1;
        if ((threadIdx.x & 15) == 0) toks[row] = tok;
        if (tok >= 0) {
            const float4* src = (const float4*)(x + (size_t)tok * Hc + c0);
#pragma unroll
            for (int j = 0; j < 8; ++j) {
                float4 a = src[2 * j], b = src[2 * j + 1];
                bf16x8 v;
                v[0] = (__bf16)a.x; v[1] = (__bf16)a.y;
                v[2] = (__bf16)a.z; v[3] = (__bf16)a.w;
                v[4] = (__bf16)b.x; v[5] = (__bf16)b.y;
                v[6] = (__bf16)b.z; v[7] = (__bf16)b.w;
                *(bf16x8*)(xs + row * XS_STRIDE + c0 + j * 8) = v;
            }
        } else {
            bf16x8 z = {};
#pragma unroll
            for (int j = 0; j < 8; ++j)
                *(bf16x8*)(xs + row * XS_STRIDE + c0 + j * 8) = z;
        }
    }
    __syncthreads();

    const int wv   = threadIdx.x >> 5;   // wave id, 0..7
    const int l    = threadIdx.x & 31;   // lane
    const int half = l >> 4;             // lane half (K-split for A/B frags)
    const int rA   = l & 15;             // A row == B/C column

    // ---- Stage A: gate & up projections + gelu*up -> act LDS ----
    {
        const float* wg = w_gate + (size_t)e * Ic * Hc;
        const float* wu = w_up   + (size_t)e * Ic * Hc;
        const __bf16* xrow = xs + rA * XS_STRIDE + half * 16;
        for (int it = wv; it < Ic / 16; it += 8) {
            const int i0 = it * 16;
            v8f cg = {0, 0, 0, 0, 0, 0, 0, 0};
            v8f cu = {0, 0, 0, 0, 0, 0, 0, 0};
            const float* wgp = wg + (size_t)(i0 + rA) * Hc + half * 16;
            const float* wup = wu + (size_t)(i0 + rA) * Hc + half * 16;
            for (int k0 = 0; k0 < Hc; k0 += 32) {
                // keep the weight streams ~1KB ahead of the WMMA consume loop
                __builtin_prefetch(wgp + k0 + 256, 0, 0);
                __builtin_prefetch(wup + k0 + 256, 0, 0);
                v16bf a  = load_a_frag(xrow, k0);
                v16bf bg = load_b_frag(wgp + k0);
                v16bf bu = load_b_frag(wup + k0);
                cg = wmma_bf16(a, bg, cg);
                cu = wmma_bf16(a, bu, cu);
            }
#pragma unroll
            for (int r = 0; r < 8; ++r) {
                const int m = r + half * 8;   // C-frag row (ISA layout)
                acts[m * AS_STRIDE + i0 + rA] = (__bf16)(gelu_tanh(cg[r]) * cu[r]);
            }
        }
    }
    __syncthreads();

    // ---- Stage B: down projection, scatter fp32 rows to output ----
    {
        const float* wd = w_down + (size_t)e * Hc * Ic;
        const __bf16* arow = acts + rA * AS_STRIDE + half * 16;
        for (int ht = wv; ht < Hc / 16; ht += 8) {
            const int h0 = ht * 16;
            v8f c = {0, 0, 0, 0, 0, 0, 0, 0};
            const float* wdp = wd + (size_t)(h0 + rA) * Ic + half * 16;
            for (int k0 = 0; k0 < Ic; k0 += 32) {
                __builtin_prefetch(wdp + k0 + 256, 0, 0);
                v16bf a = load_a_frag(arow, k0);
                v16bf b = load_b_frag(wdp + k0);
                c = wmma_bf16(a, b, c);
            }
#pragma unroll
            for (int r = 0; r < 8; ++r) {
                const int m = r + half * 8;
                if (m < rem) {
                    int t = toks[m];
                    out[(size_t)t * Hc + h0 + rA] = c[r];
                }
            }
        }
    }
}

extern "C" void kernel_launch(void* const* d_in, const int* in_sizes, int n_in,
                              void* d_out, int out_size, void* d_ws, size_t ws_size,
                              hipStream_t stream) {
    const float* x      = (const float*)d_in[0];  // [B,S,H]
    const float* gate_w = (const float*)d_in[1];  // [E,H]
    const float* w_gate = (const float*)d_in[2];  // [E,I,H]
    const float* w_up   = (const float*)d_in[3];  // [E,I,H]
    const float* w_down = (const float*)d_in[4];  // [E,H,I]

    float* out    = (float*)d_out;                  // [B*S*H]
    float* logits = out + (size_t)Bc * Sc * Hc;     // [T*E]

    int* cnt  = (int*)d_ws;                         // E counters
    int* perm = (int*)((char*)d_ws + 256);          // E*T token ids

    hipMemsetAsync(cnt, 0, Ec * sizeof(int), stream);
    router_kernel<<<Tc / 256, 256, 0, stream>>>(x, gate_w, logits, cnt, perm);
    moe_expert_kernel<<<Ec * MAX_TILES, 256, SMEM_BYTES, stream>>>(
        x, w_gate, w_up, w_down, out, cnt, perm);
}
